// One_Sent2Other_Sent_61744449847445
// MI455X (gfx1250) — matrix operand order repaired and checked by reading, not compiled, not run on hardware
//
#include <hip/hip_runtime.h>
#include <hip/hip_bf16.h>
#include <math.h>

// ---------------------------------------------------------------------------
// Problem constants (reference: N=512, D=256)
// ---------------------------------------------------------------------------
#define NROWS 512
#define DDIM  256
#define NEG_BIG (-3.402823466e38f)

typedef __attribute__((ext_vector_type(16))) __bf16 v16bf;
typedef __attribute__((ext_vector_type(8)))  float  v8f;

// |d0|,|d1| -> packed bf16x2 (lo = d0), truncation rounding.
// Pack raw diffs with one v_perm_b32, then clear BOTH bf16 sign bits with a
// single packed AND (|trunc(x)| == trunc(|x|) for sign-magnitude formats).
__device__ __forceinline__ unsigned int pack_abs_bf16x2(float d0, float d1) {
#if __has_builtin(__builtin_amdgcn_perm)
    unsigned int p = __builtin_amdgcn_perm(__float_as_uint(d1), __float_as_uint(d0),
                                           0x07060302u);
    return p & 0x7fff7fffu;
#else
    unsigned int ul = __float_as_uint(fabsf(d0));
    unsigned int uh = __float_as_uint(fabsf(d1));
    return (ul >> 16) | (uh & 0xffff0000u);
#endif
}

__device__ __forceinline__ unsigned short f32_to_bf16_bits(float x) {
    unsigned int u = __float_as_uint(x);
    u += 0x7fffu + ((u >> 16) & 1u);
    return (unsigned short)(u >> 16);
}

// ---------------------------------------------------------------------------
// Stage 0a: selfb[i,k] = X[i,:]·W1[k,:] + b_pair[k] ; other_t[i,k] = X[i,:]·W2[k,:]
// ---------------------------------------------------------------------------
__global__ __launch_bounds__(256) void selfother_kernel(
    const float* __restrict__ X, const float* __restrict__ W_pair,
    const float* __restrict__ b_pair,
    float* __restrict__ selfb, float* __restrict__ other_t)
{
    int idx = blockIdx.x * 256 + threadIdx.x;       // i*256 + k
    int i = idx >> 8, k = idx & 255;
    const float* xr = X + i * DDIM;
    const float* w1 = W_pair + k * (3 * DDIM);
    const float* w2 = w1 + DDIM;
    float s1 = 0.f, s2 = 0.f;
#pragma unroll 8
    for (int d = 0; d < DDIM; ++d) {
        float x = xr[d];
        s1 += x * w1[d];
        s2 += x * w2[d];
    }
    selfb[idx]   = s1 + b_pair[k];
    other_t[idx] = s2;
}

// ---------------------------------------------------------------------------
// Stage 0b: pre-swizzle W3 (W3[k,d] = W_pair[k*768 + 512 + d]) into bf16 WMMA
// B-fragment order (B is 32x16, K=d-step, N=k): lanes 0-15 hold K=0..15,
// lanes 16-31 hold K=16..31; fragment element idx = K%16.
// Linear: w3frag[((kt*8 + s)*32 + lane)*16 + idx]
// ---------------------------------------------------------------------------
__global__ __launch_bounds__(256) void w3pack_kernel(
    const float* __restrict__ W_pair, unsigned short* __restrict__ w3frag)
{
    int t = blockIdx.x * 256 + threadIdx.x;         // (kt*8 + s)*32 + lane, 0..4095
    int lane = t & 31;
    int s    = (t >> 5) & 7;
    int kt   = t >> 8;                              // 0..15
    int jl   = lane & 15;
    int half = lane >> 4;
    int k    = kt * 16 + jl;

    unsigned short outv[16];
#pragma unroll
    for (int idx = 0; idx < 16; ++idx) {
        int d = 32 * s + 16 * half + idx;
        outv[idx] = f32_to_bf16_bits(W_pair[k * (3 * DDIM) + 2 * DDIM + d]);
    }
    uint4* dst = (uint4*)(w3frag + (size_t)t * 16);
    dst[0] = ((const uint4*)outv)[0];
    dst[1] = ((const uint4*)outv)[1];
}

// ---------------------------------------------------------------------------
// Main fused kernel: one block per i (256 threads = 8 waves).
//   pre[j,k] = sum_d bf16(|X[i,d]-X[j,d]|) * bf16(W3[k,d])  (WMMA f32 acc)
//              + other_t[j,k]
//   pooled[i,k] = relu( max_{j in mask} pre[j,k] + selfb[i,k] )
//
// Wave w: k-group kg = w&3 (k-tiles 4kg..4kg+3), j-half jh = w>>2.
// Each step t stages TWO j-tiles (t and 16+t); wave group jh consumes slot jh.
// Each wave builds ONE A fragment per K-step and feeds 4 WMMAs with it.
// s-loop kept as a rolled loop (unroll 2) so B fragments stay s-indexed and
// cannot be hoisted across the t-loop (round-2 showed that hoisting all 32
// fragments spills past 256 VGPRs).
// ---------------------------------------------------------------------------
__global__ __launch_bounds__(256) void pair_gemm_pool_kernel(
    const float* __restrict__ X, const float* __restrict__ selfb,
    const float* __restrict__ other_t, const unsigned short* __restrict__ w3frag,
    float* __restrict__ pooled)
{
    __shared__ float xi[DDIM];           //  1 KB : row i of X
    __shared__ float xj2[2][16][DDIM];   // 32 KB : two staged X j-tiles
    __shared__ float ot2[2][16][DDIM];   // 32 KB : two staged other_t j-tiles
    __shared__ float pmax[2][DDIM];      //  2 KB : per-j-half partial maxes

    const int i    = blockIdx.x;
    const int tid  = threadIdx.x;
    const int lane = tid & 31;
    const int wave = tid >> 5;           // 0..7
    const int kg   = wave & 3;           // k-group: k-tiles 4kg..4kg+3
    const int jh   = wave >> 2;          // j-half: j-tiles jh*16 .. jh*16+15
    const int jl   = lane & 15;
    const int half = lane >> 4;

    xi[tid] = X[(size_t)i * DDIM + tid];

    float pm0 = NEG_BIG, pm1 = NEG_BIG, pm2 = NEG_BIG, pm3 = NEG_BIG;
    const int k0 = (kg * 4 + 0) * 16 + jl;
    const int k1 = (kg * 4 + 1) * 16 + jl;
    const int k2 = (kg * 4 + 2) * 16 + jl;
    const int k3 = (kg * 4 + 3) * 16 + jl;

    // per-wave base of its 32 B fragments: ((kt*8 + s)*32 + lane)*16 halves
    const unsigned short* bbase = w3frag + (size_t)(kg * 4) * 8 * 32 * 16 + (size_t)lane * 16;

    for (int t = 0; t < 16; ++t) {
        __syncthreads();   // previous tiles fully consumed (and xi visible on t==0)
        {
            // stage j-tiles t (slot 0) and 16+t (slot 1): 4 x 1024 float4
            const float4* gx0 = (const float4*)(X       + (size_t)(t * 16) * DDIM);
            const float4* gx1 = (const float4*)(X       + (size_t)((16 + t) * 16) * DDIM);
            const float4* go0 = (const float4*)(other_t + (size_t)(t * 16) * DDIM);
            const float4* go1 = (const float4*)(other_t + (size_t)((16 + t) * 16) * DDIM);
            float4* sx0 = (float4*)&xj2[0][0][0];
            float4* sx1 = (float4*)&xj2[1][0][0];
            float4* so0 = (float4*)&ot2[0][0][0];
            float4* so1 = (float4*)&ot2[1][0][0];
#pragma unroll
            for (int q = 0; q < 4; ++q) {
                sx0[tid + 256 * q] = gx0[tid + 256 * q];
                sx1[tid + 256 * q] = gx1[tid + 256 * q];
                so0[tid + 256 * q] = go0[tid + 256 * q];
                so1[tid + 256 * q] = go1[tid + 256 * q];
            }
        }
        __syncthreads();

        const int jt = jh * 16 + t;
        v8f c0 = {}, c1 = {}, c2 = {}, c3 = {};
#pragma unroll 2
        for (int s = 0; s < 8; ++s) {
            // ---- B fragments first: global (L2-hot), two coalesced 16B loads
            //      each; latency overlaps the A-build VALU chain below ----
            union { uint4 q[2]; v16bf v; } b0, b1, b2, b3;
            const uint4* p0 = (const uint4*)(bbase + (size_t)((0 * 8 + s) * 32) * 16);
            const uint4* p1 = (const uint4*)(bbase + (size_t)((1 * 8 + s) * 32) * 16);
            const uint4* p2 = (const uint4*)(bbase + (size_t)((2 * 8 + s) * 32) * 16);
            const uint4* p3 = (const uint4*)(bbase + (size_t)((3 * 8 + s) * 32) * 16);
            b0.q[0] = p0[0]; b0.q[1] = p0[1];
            b1.q[0] = p1[0]; b1.q[1] = p1[1];
            b2.q[0] = p2[0]; b2.q[1] = p2[1];
            b3.q[0] = p3[0]; b3.q[1] = p3[1];

            // ---- A fragment: 16x32 bf16 of |X[j,d]-X[i,d]| ----
            // lanes 0-15 / 16-31 both rows M=0..15; VGPR v: K=16*half+2v(+1)
            union { unsigned int u[8]; v16bf v; } af;
            const float* xjrow = &xj2[jh][jl][0];
#pragma unroll
            for (int v = 0; v < 8; ++v) {
                int d = 32 * s + 16 * half + 2 * v;
                float2 a = *(const float2*)&xjrow[d];
                float2 b = *(const float2*)&xi[d];
                af.u[v] = pack_abs_bf16x2(a.x - b.x, a.y - b.y);
            }

            c0 = __builtin_amdgcn_wmma_f32_16x16x32_bf16(false, af.v, false, b0.v, (short)0, c0, false, false);
            c1 = __builtin_amdgcn_wmma_f32_16x16x32_bf16(false, af.v, false, b1.v, (short)0, c1, false, false);
            c2 = __builtin_amdgcn_wmma_f32_16x16x32_bf16(false, af.v, false, b2.v, (short)0, c2, false, false);
            c3 = __builtin_amdgcn_wmma_f32_16x16x32_bf16(false, af.v, false, b3.v, (short)0, c3, false, false);
        }

        // ---- post-process tile: + other_t, mask, running max over j ----
        // C layout: VGPR r, lanes 0-15: M=r, N=lane; lanes 16-31: M=r+8, N=lane-16
#pragma unroll
        for (int r = 0; r < 8; ++r) {
            int jloc = r + 8 * half;
            int j    = jt * 16 + jloc;
            bool dead = (j == i) || ((i == NROWS - 1) && (j == NROWS - 2));
            float v0 = c0[r] + ot2[jh][jloc][k0];
            float v1 = c1[r] + ot2[jh][jloc][k1];
            float v2 = c2[r] + ot2[jh][jloc][k2];
            float v3 = c3[r] + ot2[jh][jloc][k3];
            if (dead) { v0 = NEG_BIG; v1 = NEG_BIG; v2 = NEG_BIG; v3 = NEG_BIG; }
            pm0 = fmaxf(pm0, v0);
            pm1 = fmaxf(pm1, v1);
            pm2 = fmaxf(pm2, v2);
            pm3 = fmaxf(pm3, v3);
        }
    }

    // combine the two lane-halves (each covered a disjoint set of 8 rows/tile)
    pm0 = fmaxf(pm0, __shfl_xor(pm0, 16, 32));
    pm1 = fmaxf(pm1, __shfl_xor(pm1, 16, 32));
    pm2 = fmaxf(pm2, __shfl_xor(pm2, 16, 32));
    pm3 = fmaxf(pm3, __shfl_xor(pm3, 16, 32));

    if (half == 0) {
        pmax[jh][k0] = pm0;
        pmax[jh][k1] = pm1;
        pmax[jh][k2] = pm2;
        pmax[jh][k3] = pm3;
    }
    __syncthreads();

    // combine the two j-halves; fold in self term + bias, ReLU, store
    {
        float m = fmaxf(pmax[0][tid], pmax[1][tid]);
        pooled[(size_t)i * DDIM + tid] = fmaxf(m + selfb[(size_t)i * DDIM + tid], 0.f);
    }
}

// ---------------------------------------------------------------------------
// Stage 2: gates -> hid.  gates = pooled @ W_ih.T + b_ih + b_hh (W_hh * 0 drops)
// ---------------------------------------------------------------------------
__global__ __launch_bounds__(256) void gates_kernel(
    const float* __restrict__ pooled, const float* __restrict__ W_ih,
    const float* __restrict__ b_ih, const float* __restrict__ b_hh,
    float* __restrict__ hid)
{
    int idx = blockIdx.x * 256 + threadIdx.x;       // n*256 + k
    int n = idx >> 8, k = idx & 255;
    const float* p  = pooled + (size_t)n * DDIM;
    const float* wi = W_ih + (size_t)(k)       * DDIM;  // row k     (i-gate)
    const float* wg = W_ih + (size_t)(512 + k) * DDIM;  // row 2d+k  (g-gate)
    const float* wo = W_ih + (size_t)(768 + k) * DDIM;  // row 3d+k  (o-gate)
    float si = 0.f, sg = 0.f, so = 0.f;
#pragma unroll 4
    for (int d = 0; d < DDIM; ++d) {
        float x = p[d];
        si += x * wi[d];
        sg += x * wg[d];
        so += x * wo[d];
    }
    si += b_ih[k]       + b_hh[k];
    sg += b_ih[512 + k] + b_hh[512 + k];
    so += b_ih[768 + k] + b_hh[768 + k];
    float ig = 1.f / (1.f + __expf(-si));
    float gg = tanhf(sg);
    float og = 1.f / (1.f + __expf(-so));
    hid[idx] = og * tanhf(ig * gg);
}

// ---------------------------------------------------------------------------
// Stage 3: attn[n] = hid[n,:]·W_attn + b_attn;  pooled2[k] = sum_n hid[n,k]*attn[n]
//          out = sigmoid(pooled2·W_fc + b_fc)   (single scalar)
// ---------------------------------------------------------------------------
__global__ __launch_bounds__(512) void final_kernel(
    const float* __restrict__ hid, const float* __restrict__ W_attn,
    const float* __restrict__ b_attn, const float* __restrict__ W_fc,
    const float* __restrict__ b_fc, float* __restrict__ out)
{
    __shared__ float attn[NROWS];
    __shared__ float red[256];
    int tid = threadIdx.x;

    {   // phase 1: per-row attention scalar
        float s = 0.f;
        const float* h = hid + (size_t)tid * DDIM;
#pragma unroll 8
        for (int d = 0; d < DDIM; ++d) s += h[d] * W_attn[d];
        attn[tid] = s + b_attn[0];
    }
    __syncthreads();

    // phase 2: weighted column sums * W_fc, block reduction
    if (tid < 256) {
        float p2 = 0.f;
        for (int n = 0; n < NROWS; ++n) p2 += hid[(size_t)n * DDIM + tid] * attn[n];
        red[tid] = p2 * W_fc[tid];
    }
    __syncthreads();
    for (int off = 128; off > 0; off >>= 1) {
        if (tid < off) red[tid] += red[tid + off];
        __syncthreads();
    }
    if (tid == 0) out[0] = 1.f / (1.f + __expf(-(red[0] + b_fc[0])));
}

// ---------------------------------------------------------------------------
// Launcher
// ---------------------------------------------------------------------------
extern "C" void kernel_launch(void* const* d_in, const int* in_sizes, int n_in,
                              void* d_out, int out_size, void* d_ws, size_t ws_size,
                              hipStream_t stream) {
    const float* X      = (const float*)d_in[0];   // (512, 256)
    const float* W_pair = (const float*)d_in[1];   // (256, 768)
    const float* b_pair = (const float*)d_in[2];   // (256,)
    const float* W_ih   = (const float*)d_in[3];   // (1024, 256)
    // d_in[4] = W_hh: multiplied by zeros in the reference -> unused
    const float* b_ih   = (const float*)d_in[5];   // (1024,)
    const float* b_hh   = (const float*)d_in[6];   // (1024,)
    const float* W_attn = (const float*)d_in[7];   // (1, 256)
    const float* b_attn = (const float*)d_in[8];   // (1,)
    const float* W_fc   = (const float*)d_in[9];   // (1, 256)
    const float* b_fc   = (const float*)d_in[10];  // (1,)
    float* out = (float*)d_out;

    char* ws = (char*)d_ws;
    float*          selfb   = (float*)(ws);                        // 512 KB
    float*          other_t = (float*)(ws +  512 * 1024);          // 512 KB
    float*          pooled  = (float*)(ws + 1024 * 1024);          // 512 KB
    float*          hid     = (float*)(ws + 1536 * 1024);          // 512 KB
    unsigned short* w3frag  = (unsigned short*)(ws + 2048 * 1024); // 128 KB

    selfother_kernel<<<NROWS, 256, 0, stream>>>(X, W_pair, b_pair, selfb, other_t);
    w3pack_kernel<<<16, 256, 0, stream>>>(W_pair, w3frag);
    pair_gemm_pool_kernel<<<NROWS, 256, 0, stream>>>(X, selfb, other_t, w3frag, pooled);
    gates_kernel<<<NROWS, 256, 0, stream>>>(pooled, W_ih, b_ih, b_hh, hid);
    final_kernel<<<1, 512, 0, stream>>>(hid, W_attn, b_attn, W_fc, b_fc, out);
}
